// TransformerBlock_39702677684934
// MI455X (gfx1250) — matrix operand order repaired and checked by reading, not compiled, not run on hardware
//
#include <hip/hip_runtime.h>
#include <hip/hip_bf16.h>

// Problem constants (from the reference)
#define B_    2
#define C_    64
#define H_    32
#define W_    32
#define D_    32
#define N_    32768      // H*W*D tokens
#define P_    64
#define NH_   4
#define HD_   16
#define QKVV_ 256
#define LN_EPS 1e-5f
#define BN_EPS 1e-5f

typedef __attribute__((ext_vector_type(16))) _Float16 v16h;
typedef __attribute__((ext_vector_type(8)))  _Float16 v8h;
typedef __attribute__((ext_vector_type(8)))  float    v8f;
typedef __attribute__((ext_vector_type(4)))  float    v4f;

__device__ __forceinline__ int laneid() { return (int)(threadIdx.x & 31u); }

__device__ __forceinline__ v8f wmma_f16(v16h a, v16h b, v8f c) {
  // D = A(16x32 f16) * B(32x16 f16) + C(16x16 f32)
  return __builtin_amdgcn_wmma_f32_16x16x32_f16(false, a, false, b, (short)0, c,
                                                false, false);
}

__device__ __forceinline__ v8f zero8() {
  v8f z;
#pragma unroll
  for (int i = 0; i < 8; ++i) z[i] = 0.f;
  return z;
}

// ---- fragment gathers per CDNA5 ISA 7.12.2 (wave32) ----
// A (16x32 f16): lanes 0-15 row M=lane hold K={0..7,16..23};
//                lanes 16-31 row M=lane-16 hold K={8..15,24..31}.
// generic strided gather: element (m,k) at p[m*ms + k*ks]
__device__ __forceinline__ v16h gatherA_h(const _Float16* p, size_t ms, size_t ks) {
  int l = laneid(); int m = l & 15; int kb = (l < 16) ? 0 : 8;
  const _Float16* r = p + (size_t)m * ms;
  v16h a;
#pragma unroll
  for (int i = 0; i < 8; ++i) {
    a[i]     = r[(size_t)(kb + i) * ks];
    a[i + 8] = r[(size_t)(kb + 16 + i) * ks];
  }
  return a;
}

// A fast path: row-major A (k-stride 1) -> two 16B loads per lane
__device__ __forceinline__ v16h loadA_rm(const _Float16* p, size_t ms) {
  int l = laneid(); int m = l & 15; int kb = (l < 16) ? 0 : 8;
  const _Float16* r = p + (size_t)m * ms + kb;
  v8h lo = *(const v8h*)r;          // K = kb .. kb+7
  v8h hi = *(const v8h*)(r + 16);   // K = kb+16 .. kb+23
  v16h a;
#pragma unroll
  for (int i = 0; i < 8; ++i) { a[i] = lo[i]; a[i + 8] = hi[i]; }
  return a;
}

// B (32x16 f16): lanes 0-15 col N=lane hold K={0..7,16..23}; lanes 16-31 hold K={8..15,24..31}.
// generic strided gather: element (k,n) at p[k*ks + n*ns]
__device__ __forceinline__ v16h gatherB_h(const _Float16* p, size_t ks, size_t ns) {
  int l = laneid(); int n = l & 15; int kb = (l < 16) ? 0 : 8;
  const _Float16* r = p + (size_t)n * ns;
  v16h bv;
#pragma unroll
  for (int i = 0; i < 16; ++i) bv[i] = r[(size_t)(kb + (i < 8 ? i : 8 + i)) * ks];
  return bv;
}

// B fast path: [n][k] layout (k contiguous) -> two 16B loads per lane
__device__ __forceinline__ v16h loadB_nm(const _Float16* p, size_t ns) {
  int l = laneid(); int n = l & 15; int kb = (l < 16) ? 0 : 8;
  const _Float16* r = p + (size_t)n * ns + kb;
  v8h lo = *(const v8h*)r;
  v8h hi = *(const v8h*)(r + 16);
  v16h bv;
#pragma unroll
  for (int i = 0; i < 8; ++i) { bv[i] = lo[i]; bv[i + 8] = hi[i]; }
  return bv;
}

// B fast path from f32 [n][k] (k contiguous): four float4 loads + cvt
__device__ __forceinline__ v16h loadB_nm_f32(const float* p, size_t ns) {
  int l = laneid(); int n = l & 15; int kb = (l < 16) ? 0 : 8;
  const float* r = p + (size_t)n * ns + kb;
  v4f x0 = *(const v4f*)(r);
  v4f x1 = *(const v4f*)(r + 4);
  v4f x2 = *(const v4f*)(r + 16);
  v4f x3 = *(const v4f*)(r + 20);
  v16h bv;
#pragma unroll
  for (int i = 0; i < 4; ++i) {
    bv[i]      = (_Float16)x0[i];
    bv[i + 4]  = (_Float16)x1[i];
    bv[i + 8]  = (_Float16)x2[i];
    bv[i + 12] = (_Float16)x3[i];
  }
  return bv;
}

// D/C (16x16 f32): VGPR j: lanes 0-15 -> (M=j, N=lane); lanes 16-31 -> (M=8+j, N=lane-16)

// ============================================================================
// K1: xs transpose + LayerNorm(C=64) + qkvv = xn @ W_qkvv^T  (f16, [B*N,256])
// ============================================================================
__global__ void k_ln_qkvv(const float* __restrict__ x, const float* __restrict__ lnw,
                          const float* __restrict__ lnb, const float* __restrict__ Wq,
                          _Float16* __restrict__ q16) {
  __shared__ float Xs[16][64];
  __shared__ alignas(16) _Float16 An[16][64];
  const int r0  = blockIdx.x * 16;     // global row = b*N + n
  const int b   = r0 / N_;
  const int n0  = r0 % N_;
  const int tid = threadIdx.x;
  for (int idx = tid; idx < 16 * 64; idx += 32) {
    int row = idx >> 6, c = idx & 63;
    Xs[row][c] = x[(size_t)b * C_ * N_ + (size_t)c * N_ + n0 + row];
  }
  __syncthreads();
  if (tid < 16) {
    float mu = 0.f;
    for (int c = 0; c < 64; ++c) mu += Xs[tid][c];
    mu *= (1.f / 64.f);
    float var = 0.f;
    for (int c = 0; c < 64; ++c) { float d = Xs[tid][c] - mu; var += d * d; }
    var *= (1.f / 64.f);
    float rs = rsqrtf(var + LN_EPS);
    for (int c = 0; c < 64; ++c)
      An[tid][c] = (_Float16)((Xs[tid][c] - mu) * rs * lnw[c] + lnb[c]);
  }
  __syncthreads();
  v16h a0 = loadA_rm(&An[0][0],  64);
  v16h a1 = loadA_rm(&An[0][32], 64);
  const int l = laneid();
  for (int ot = 0; ot < 16; ++ot) {
    int o0 = ot * 16;
    v8f acc = zero8();
    acc = wmma_f16(a0, loadB_nm_f32(Wq + (size_t)o0 * 64,      64), acc);
    acc = wmma_f16(a1, loadB_nm_f32(Wq + (size_t)o0 * 64 + 32, 64), acc);
#pragma unroll
    for (int j = 0; j < 8; ++j) {
      int rr = (l < 16) ? j : 8 + j;
      int cc = l & 15;
      q16[(size_t)(r0 + rr) * QKVV_ + o0 + cc] = (_Float16)acc[j];
    }
  }
}

// ============================================================================
// K2: 1/max(||q_d||,1e-12) and same for k, over the token axis N
// ============================================================================
__global__ void k_qk_norm(const _Float16* __restrict__ q16, float* __restrict__ invn) {
  __shared__ float red[256];
  const int bi   = blockIdx.x;
  const int comp = bi >> 3;           // 0=q, 1=k
  const int b    = (bi >> 2) & 1;
  const int h    = bi & 3;
  const int tid  = threadIdx.x;
  const int chOff = comp * 64 + h * 16;
  float s[16];
#pragma unroll
  for (int d = 0; d < 16; ++d) s[d] = 0.f;
  for (int n = tid; n < N_; n += 256) {
    const _Float16* p = q16 + (size_t)(b * N_ + n) * QKVV_ + chOff;
    v8h x0 = *(const v8h*)p;
    v8h x1 = *(const v8h*)(p + 8);
#pragma unroll
    for (int d = 0; d < 8; ++d) {
      float v0 = (float)x0[d]; s[d]     += v0 * v0;
      float v1 = (float)x1[d]; s[d + 8] += v1 * v1;
    }
  }
  for (int d = 0; d < 16; ++d) {
    red[tid] = s[d];
    __syncthreads();
    for (int o = 128; o > 0; o >>= 1) {
      if (tid < o) red[tid] += red[tid + o];
      __syncthreads();
    }
    if (tid == 0)
      invn[((size_t)(comp * 2 + b) * 4 + h) * 16 + d] = 1.f / fmaxf(sqrtf(red[0]), 1e-12f);
    __syncthreads();
  }
}

// ============================================================================
// K3a: split-K partials of {k,v_sa} @ W_EF^T ; slice per split -> deterministic
//   kvFp layout: [split16][comp2][bh8][d16][p64] f32
// ============================================================================
__global__ void k_kvprojp(const _Float16* __restrict__ q16, const float* __restrict__ WEF,
                          float* __restrict__ kvFp) {
  const int bi   = blockIdx.x;
  const int pt   = bi & 3;
  const int h    = (bi >> 2) & 3;
  const int b    = (bi >> 4) & 1;
  const int comp = (bi >> 5) & 1;      // 0 -> k (chan 64..), 1 -> v_sa (chan 192..)
  const int sp   = bi >> 6;            // 0..15
  const int p0   = pt * 16;
  const int chOff = comp ? 192 : 64;
  v8f acc = zero8();
  for (int kt = sp * 64; kt < sp * 64 + 64; ++kt) {
    const _Float16* abase = q16 + (size_t)(b * N_ + kt * 32) * QKVV_ + chOff;
    v16h a  = gatherA_h(abase, 1, QKVV_);                      // A[d][n]
    v16h bb = loadB_nm_f32(WEF + (size_t)p0 * N_ + kt * 32, N_); // B[n][p]=WEF[p][n]
    acc = wmma_f16(a, bb, acc);
  }
  const int l = laneid();
  float* outp = kvFp + (size_t)sp * 16384 + (size_t)comp * 8192 +
                (size_t)(b * 4 + h) * 1024;
#pragma unroll
  for (int j = 0; j < 8; ++j) {
    int rr = (l < 16) ? j : 8 + j;
    int cc = l & 15;
    outp[(size_t)rr * 64 + p0 + cc] = acc[j];
  }
}

// K3b: fixed-order reduction + bias; k stored transposed [bh][p][d] for k_sa
__global__ void k_kvpack(const float* __restrict__ kvFp, const float* __restrict__ bEF,
                         _Float16* __restrict__ kprojT, _Float16* __restrict__ vproj) {
  int i = blockIdx.x * 256 + threadIdx.x;   // 16384 outputs
  if (i >= 16384) return;
  float s = 0.f;
#pragma unroll
  for (int sp = 0; sp < 16; ++sp) s += kvFp[(size_t)sp * 16384 + i];
  int comp = i >> 13;
  int rem  = i & 8191;                  // [bh8][d16][p64]
  int p  = rem & 63;
  int d  = (rem >> 6) & 15;
  int bh = rem >> 10;
  float v = s + bEF[p];
  if (comp == 0) kprojT[((size_t)bh * 64 + p) * 16 + d] = (_Float16)v;
  else           vproj [((size_t)bh * 16 + d) * 64 + p] = (_Float16)v;
}

// ============================================================================
// K4a: split-K partials of the q@k^T Gram matrix; [split32][bh8][16][16] f32
// ============================================================================
__global__ void k_gramp(const _Float16* __restrict__ q16, float* __restrict__ gramp) {
  const int bi = blockIdx.x;
  const int h  = bi & 3;
  const int b  = (bi >> 2) & 1;
  const int sp = bi >> 3;               // 0..31
  v8f acc = zero8();
  for (int kt = sp * 32; kt < sp * 32 + 32; ++kt) {
    const _Float16* base = q16 + (size_t)(b * N_ + kt * 32) * QKVV_;
    v16h a  = gatherA_h(base + h * 16,      1,     QKVV_);  // q: A[d][n]
    v16h bb = gatherB_h(base + 64 + h * 16, QKVV_, 1);      // k^T: B[n][e]
    acc = wmma_f16(a, bb, acc);
  }
  const int l = laneid();
#pragma unroll
  for (int j = 0; j < 8; ++j) {
    int rr = (l < 16) ? j : 8 + j;
    int cc = l & 15;
    gramp[(size_t)sp * 2048 + ((size_t)(b * 4 + h) * 16 + rr) * 16 + cc] = acc[j];
  }
}

// K4b: reduce + l2norm scaling + temp1 + softmax -> attn_ca [B,NH,16,16] f32
__global__ void k_attnfin(const float* __restrict__ gramp, const float* __restrict__ invn,
                          const float* __restrict__ temp1, float* __restrict__ attnca) {
  const int b = blockIdx.x >> 2, h = blockIdx.x & 3;
  if (threadIdx.x >= 16) return;
  const int r  = threadIdx.x;
  const float t1 = temp1[h];
  const float iq = invn[((0 * 2 + b) * 4 + h) * 16 + r];
  float row[16];
  for (int e = 0; e < 16; ++e) {
    float s = 0.f;
    for (int sp = 0; sp < 32; ++sp)
      s += gramp[(size_t)sp * 2048 + ((size_t)(b * 4 + h) * 16 + r) * 16 + e];
    row[e] = s * iq * invn[((1 * 2 + b) * 4 + h) * 16 + e] * t1;
  }
  float m = row[0];
  for (int e = 1; e < 16; ++e) m = fmaxf(m, row[e]);
  float sum = 0.f, ex[16];
  for (int e = 0; e < 16; ++e) { ex[e] = __expf(row[e] - m); sum += ex[e]; }
  float inv = 1.f / sum;
  for (int e = 0; e < 16; ++e)
    attnca[((size_t)(b * 4 + h) * 16 + r) * 16 + e] = ex[e] * inv;
}

// ============================================================================
// K5: x_ca = attn_ca @ v_ca, permuted to [B,N,C] (c = h*16+d), f16
// ============================================================================
__global__ void k_xca(const _Float16* __restrict__ q16, const float* __restrict__ attnca,
                      _Float16* __restrict__ xca) {
  __shared__ float Att[NH_][16][16];
  const int b = blockIdx.x >> 7;
  const int n = ((blockIdx.x & 127) << 8) + threadIdx.x;
  for (int idx = threadIdx.x; idx < NH_ * 256; idx += 256)
    ((float*)Att)[idx] = attnca[(size_t)b * NH_ * 256 + idx];
  __syncthreads();
  const _Float16* vp = q16 + (size_t)(b * N_ + n) * QKVV_ + 128;   // v_ca channels
  float vv[64];
#pragma unroll
  for (int j = 0; j < 8; ++j) {
    v8h t = *(const v8h*)(vp + j * 8);
#pragma unroll
    for (int e = 0; e < 8; ++e) vv[j * 8 + e] = (float)t[e];
  }
  _Float16* outp = xca + (size_t)(b * N_ + n) * 64;
  for (int h = 0; h < NH_; ++h)
    for (int d = 0; d < 16; ++d) {
      float s = 0.f;
#pragma unroll
      for (int e = 0; e < 16; ++e) s += Att[h][d][e] * vv[h * 16 + e];
      outp[h * 16 + d] = (_Float16)s;
    }
}

// ============================================================================
// K6: spatial attention + x_sa.  Per (b,h,16-token tile):
//   S = (qn^T @ k_proj) * temp2 -> softmax over P=64 -> @ v_proj^T
//   write x_sa into the "faithful bug" layout [B][d][h][N]
// ============================================================================
__global__ void k_sa(const _Float16* __restrict__ q16, const float* __restrict__ invn,
                     const _Float16* __restrict__ kprojT, const _Float16* __restrict__ vproj,
                     const float* __restrict__ temp2, _Float16* __restrict__ xsa) {
  __shared__ float Sm[16][64];
  __shared__ alignas(16) _Float16 Pm[16][64];
  const int bi = blockIdx.x;
  const int nt = bi & 2047;
  const int h  = (bi >> 11) & 3;
  const int b  = bi >> 13;
  const int n0 = nt * 16;
  const int l  = laneid();
  const int m  = l & 15;
  const int cc = l & 15;
  // A: rows = tokens, K = head dim (16 real, slots for K>=16 zeroed)
  v16h a;
  {
    int kb = (l < 16) ? 0 : 8;
    const _Float16* qp = q16 + (size_t)(b * N_ + n0 + m) * QKVV_ + h * 16;
    const float* iqp = invn + ((size_t)(0 * 2 + b) * 4 + h) * 16 + kb;
    v8h qv = *(const v8h*)(qp + kb);
#pragma unroll
    for (int i = 0; i < 8; ++i) {
      a[i]     = (_Float16)((float)qv[i] * iqp[i]);
      a[i + 8] = (_Float16)0.f;
    }
  }
  const float t2 = temp2[h];
  const _Float16* kpT = kprojT + (size_t)(b * 4 + h) * 64 * 16;   // [p][d]
#pragma unroll
  for (int pt = 0; pt < 4; ++pt) {
    const _Float16* r = kpT + (size_t)(pt * 16 + cc) * 16;
    v8h lo = *(const v8h*)r;
    v8h hi = *(const v8h*)(r + 8);
    v16h bb;
#pragma unroll
    for (int i = 0; i < 8; ++i) {   // slots 8..15 carry K=16..31 -> zero pad
      bb[i]     = (l < 16) ? lo[i] : hi[i];
      bb[i + 8] = (_Float16)0.f;
    }
    v8f s = wmma_f16(a, bb, zero8());
#pragma unroll
    for (int j = 0; j < 8; ++j) {
      int rr = (l < 16) ? j : 8 + j;
      Sm[rr][pt * 16 + cc] = s[j] * t2;
    }
  }
  __syncthreads();
  if (threadIdx.x < 16) {
    int r = threadIdx.x;
    float mx = Sm[r][0];
    for (int p = 1; p < 64; ++p) mx = fmaxf(mx, Sm[r][p]);
    float sum = 0.f;
    for (int p = 0; p < 64; ++p) sum += __expf(Sm[r][p] - mx);
    float inv = 1.f / sum;
    for (int p = 0; p < 64; ++p) Pm[r][p] = (_Float16)(__expf(Sm[r][p] - mx) * inv);
  }
  __syncthreads();
  // x_sa tile = P(16x64) @ v_proj^T(64x16)
  v16h a0 = loadA_rm(&Pm[0][0],  64);
  v16h a1 = loadA_rm(&Pm[0][32], 64);
  const _Float16* vp = vproj + (size_t)(b * 4 + h) * 16 * 64;     // [d][p]
  v8f acc = zero8();
  acc = wmma_f16(a0, loadB_nm(vp,      64), acc);  // B[p][d] = vp[d*64+p]
  acc = wmma_f16(a1, loadB_nm(vp + 32, 64), acc);
#pragma unroll
  for (int j = 0; j < 8; ++j) {
    int rr = (l < 16) ? j : 8 + j;
    xsa[(size_t)b * (HD_ * NH_ * N_) + ((size_t)cc * NH_ + h) * N_ + n0 + rr] =
        (_Float16)acc[j];
  }
}

// ============================================================================
// K7: o-projections + gamma residual; write attn_skip f32 + f16 in [B,C,N]
// ============================================================================
__global__ void k_oproj(const float* __restrict__ x, const float* __restrict__ gamma,
                        const _Float16* __restrict__ xsa, const _Float16* __restrict__ xca,
                        const float* __restrict__ Wo1, const float* __restrict__ bo1,
                        const float* __restrict__ Wo2, const float* __restrict__ bo2,
                        float* __restrict__ askip, _Float16* __restrict__ askip16) {
  const int r0 = blockIdx.x * 16;
  const int b  = r0 / N_;
  const int n0 = r0 % N_;
  const int l  = laneid();
#pragma unroll
  for (int proj = 0; proj < 2; ++proj) {
    const _Float16* A  = (proj ? xca : xsa) + (size_t)b * N_ * 64 + (size_t)n0 * 64;
    const float*    Wp = proj ? Wo2 : Wo1;
    const float*    bp = proj ? bo2 : bo1;
    v16h a0 = loadA_rm(A,      64);
    v16h a1 = loadA_rm(A + 32, 64);
#pragma unroll
    for (int jt = 0; jt < 2; ++jt) {
      v8f acc = zero8();
      acc = wmma_f16(a0, loadB_nm_f32(Wp + (size_t)(jt * 16) * 64,      64), acc);
      acc = wmma_f16(a1, loadB_nm_f32(Wp + (size_t)(jt * 16) * 64 + 32, 64), acc);
#pragma unroll
      for (int j = 0; j < 8; ++j) {
        int rr = (l < 16) ? j : 8 + j;
        int cc = l & 15;
        int c  = proj * 32 + jt * 16 + cc;
        int n  = n0 + rr;
        size_t idx = (size_t)b * C_ * N_ + (size_t)c * N_ + n;
        float av = x[idx] + gamma[c] * (acc[j] + bp[jt * 16 + cc]);
        askip[idx]   = av;
        askip16[idx] = (_Float16)av;
      }
    }
  }
}

// ============================================================================
// conv weight prep: (O,I,3,3,3) f32 -> [sel][tap][o][ci] f16 (B-fragment friendly)
// ============================================================================
__global__ void k_wprep(const float* __restrict__ w1, const float* __restrict__ w2,
                        _Float16* __restrict__ w16) {
  int g = blockIdx.x * 256 + threadIdx.x;
  const int per = 64 * 64 * 27;
  if (g >= 2 * per) return;
  int sel = g / per;
  int r   = g % per;
  int o   = r / (64 * 27);
  int rem = r % (64 * 27);
  int ci  = rem / 27;
  int t   = rem % 27;
  const float* src = sel ? w2 : w1;
  w16[(((size_t)sel * 27 + t) * 64 + o) * 64 + ci] = (_Float16)src[r];
}

// ============================================================================
// K8/K9: 3x3x3 conv, implicit GEMM.  One wave per (b,h,w): stages the 9 input
// planes into LDS transposed to [plane][d][c] (wide b128 global loads), then
// 27 taps x 2 K-chunks x (2 d-tiles x 4 o-tiles) = 432 WMMAs with A fragments
// reused across all o-tiles.  Epilogue: BN(eval) [+skip] + LeakyReLU.
// ============================================================================
__global__ void k_conv3(const _Float16* __restrict__ in16, const _Float16* __restrict__ w16,
                        const float* __restrict__ bnw, const float* __restrict__ bnb,
                        const float* __restrict__ skip, _Float16* __restrict__ out16) {
  __shared__ alignas(16) _Float16 Xt[9][32][64];   // [plane][d][ci]
  const int bi = blockIdx.x;
  const int w  = bi & 31;
  const int h  = (bi >> 5) & 31;
  const int b  = bi >> 10;
  const int l  = laneid();
  const int tid = (int)threadIdx.x;

  // ---- stage valid input planes, transposing [ci][d] -> [d][ci] ----
  const int c = tid * 2;                 // each thread owns channel pair (c, c+1)
  for (int t0 = 0; t0 < 3; ++t0) {
    int ih = h + t0 - 1;
    if (ih < 0 || ih >= 32) continue;    // uniform
    for (int t1 = 0; t1 < 3; ++t1) {
      int iw = w + t1 - 1;
      if (iw < 0 || iw >= 32) continue;  // uniform
      int p = t0 * 3 + t1;
      const _Float16* g0 = in16 + ((((size_t)b * 64 + c)     * 32 + ih) * 32 + iw) * 32;
      const _Float16* g1 = in16 + ((((size_t)b * 64 + c + 1) * 32 + ih) * 32 + iw) * 32;
#pragma unroll
      for (int j = 0; j < 4; ++j) {
        v8h a0 = *(const v8h*)(g0 + j * 8);
        v8h a1 = *(const v8h*)(g1 + j * 8);
#pragma unroll
        for (int e = 0; e < 8; ++e) {
          union { _Float16 hh[2]; unsigned u; } pk;
          pk.hh[0] = a0[e];
          pk.hh[1] = a1[e];
          *(unsigned*)&Xt[p][j * 8 + e][c] = pk.u;
        }
      }
    }
  }
  __syncthreads();

  v8f acc[2][4];
#pragma unroll
  for (int dt = 0; dt < 2; ++dt)
#pragma unroll
    for (int ot = 0; ot < 4; ++ot) acc[dt][ot] = zero8();

  const int m   = l & 15;
  const int kbA = (l < 16) ? 0 : 8;
  const int cc  = l & 15;
  for (int t0 = 0; t0 < 3; ++t0) {
    int ih = h + t0 - 1;
    if (ih < 0 || ih >= 32) continue;
    for (int t1 = 0; t1 < 3; ++t1) {
      int iw = w + t1 - 1;
      if (iw < 0 || iw >= 32) continue;
      int p = t0 * 3 + t1;
      for (int t2 = 0; t2 < 3; ++t2) {
        int tap = p * 3 + t2;
#pragma unroll
        for (int kc = 0; kc < 2; ++kc) {
          int c0 = kc * 32;
          // B fragments for 4 o-tiles (contiguous 16B loads; [tap][o][ci])
          v16h bb[4];
#pragma unroll
          for (int ot = 0; ot < 4; ++ot)
            bb[ot] = loadB_nm(w16 + ((size_t)tap * 64 + ot * 16) * 64 + c0, 64);
#pragma unroll
          for (int dt = 0; dt < 2; ++dt) {
            int  id  = dt * 16 + m + t2 - 1;
            bool rv  = (id >= 0) && (id < 32);
            int  idc = rv ? id : 0;
            const _Float16* r = &Xt[p][idc][c0 + kbA];
            v8h lo = *(const v8h*)r;
            v8h hi = *(const v8h*)(r + 16);
            v16h a;
#pragma unroll
            for (int i = 0; i < 8; ++i) {
              a[i]     = rv ? lo[i] : (_Float16)0.f;
              a[i + 8] = rv ? hi[i] : (_Float16)0.f;
            }
#pragma unroll
            for (int ot = 0; ot < 4; ++ot) acc[dt][ot] = wmma_f16(a, bb[ot], acc[dt][ot]);
          }
        }
      }
    }
  }
#pragma unroll
  for (int dt = 0; dt < 2; ++dt)
#pragma unroll
    for (int ot = 0; ot < 4; ++ot) {
      int o = ot * 16 + cc;
      float scl = bnw[o] * rsqrtf(1.f + BN_EPS);
      float sft = bnb[o];
#pragma unroll
      for (int j = 0; j < 8; ++j) {
        int rr = (l < 16) ? j : 8 + j;
        int d  = dt * 16 + rr;
        size_t idx = ((((size_t)b * 64 + o) * 32 + h) * 32 + w) * 32 + d;
        float y = acc[dt][ot][j] * scl + sft;
        if (skip) y += skip[idx];
        y = (y >= 0.f) ? y : 0.01f * y;
        out16[idx] = (_Float16)y;
      }
    }
}

// ============================================================================
// K10: conv8 (1x1x1 GEMM) + bias + attn_skip -> final f32 output
// ============================================================================
__global__ void k_conv8(const _Float16* __restrict__ in16, const float* __restrict__ w8,
                        const float* __restrict__ b8, const float* __restrict__ askip,
                        float* __restrict__ outp) {
  const int bi = blockIdx.x;
  const int b  = bi >> 11;
  const int n0 = (bi & 2047) * 16;
  const int l  = laneid();
  const _Float16* Ab = in16 + (size_t)b * 64 * N_ + n0;
  v16h a0 = gatherA_h(Ab,                    1, N_);   // A[n][ci], ci chunk 0
  v16h a1 = gatherA_h(Ab + (size_t)32 * N_,  1, N_);   // ci chunk 1
#pragma unroll
  for (int ot = 0; ot < 4; ++ot) {
    int o0 = ot * 16;
    v8f acc = zero8();
    acc = wmma_f16(a0, loadB_nm_f32(w8 + (size_t)o0 * 64,      64), acc);
    acc = wmma_f16(a1, loadB_nm_f32(w8 + (size_t)o0 * 64 + 32, 64), acc);
    int cc = l & 15;
    int o  = o0 + cc;
#pragma unroll
    for (int j = 0; j < 8; ++j) {
      int rr = (l < 16) ? j : 8 + j;
      size_t idx = ((size_t)b * 64 + o) * N_ + n0 + rr;
      outp[idx] = askip[idx] + acc[j] + b8[o];
    }
  }
}

// ============================================================================
extern "C" void kernel_launch(void* const* d_in, const int* in_sizes, int n_in,
                              void* d_out, int out_size, void* d_ws, size_t ws_size,
                              hipStream_t stream) {
  (void)in_sizes; (void)n_in; (void)out_size; (void)ws_size;
  const float* x     = (const float*)d_in[0];
  const float* lnw   = (const float*)d_in[1];
  const float* lnb   = (const float*)d_in[2];
  const float* gamma = (const float*)d_in[3];
  const float* temp1 = (const float*)d_in[4];
  const float* temp2 = (const float*)d_in[5];
  const float* Wq    = (const float*)d_in[6];
  const float* WEF   = (const float*)d_in[7];
  const float* bEF   = (const float*)d_in[8];
  const float* Wo1   = (const float*)d_in[9];
  const float* bo1   = (const float*)d_in[10];
  const float* Wo2   = (const float*)d_in[11];
  const float* bo2   = (const float*)d_in[12];
  const float* c1w   = (const float*)d_in[13];
  const float* c2w   = (const float*)d_in[14];
  const float* bn1w  = (const float*)d_in[15];
  const float* bn1b  = (const float*)d_in[16];
  const float* bn2w  = (const float*)d_in[17];
  const float* bn2b  = (const float*)d_in[18];
  const float* c8w   = (const float*)d_in[19];
  const float* c8b   = (const float*)d_in[20];
  float* outp = (float*)d_out;

  char* ws = (char*)d_ws;
  size_t off = 0;
  auto alloc = [&](size_t bytes) -> char* {
    char* p = ws + off;
    off = (off + bytes + 255) & ~(size_t)255;
    return p;
  };
  _Float16* q16     = (_Float16*)alloc((size_t)B_ * N_ * QKVV_ * 2);     // 33.5 MB
  float*    invn    = (float*)   alloc((size_t)2 * B_ * NH_ * 16 * 4);
  float*    kvFp    = (float*)   alloc((size_t)16 * 16384 * 4);          // 1 MB
  _Float16* kprojT  = (_Float16*)alloc((size_t)B_ * NH_ * 64 * 16 * 2);
  _Float16* vproj   = (_Float16*)alloc((size_t)B_ * NH_ * 16 * 64 * 2);
  float*    gramp   = (float*)   alloc((size_t)32 * 2048 * 4);           // 256 KB
  float*    attnca  = (float*)   alloc((size_t)B_ * NH_ * 256 * 4);
  _Float16* xca     = (_Float16*)alloc((size_t)B_ * N_ * 64 * 2);        // 8.4 MB
  _Float16* xsa     = (_Float16*)alloc((size_t)B_ * N_ * 64 * 2);        // 8.4 MB
  float*    askip   = (float*)   alloc((size_t)B_ * C_ * N_ * 4);        // 16.8 MB
  _Float16* askip16 = (_Float16*)alloc((size_t)B_ * C_ * N_ * 2);        // 8.4 MB
  _Float16* out1    = (_Float16*)alloc((size_t)B_ * C_ * N_ * 2);        // 8.4 MB
  _Float16* out2    = (_Float16*)alloc((size_t)B_ * C_ * N_ * 2);        // 8.4 MB
  _Float16* w16c    = (_Float16*)alloc((size_t)2 * 27 * 64 * 64 * 2);

  k_ln_qkvv<<<dim3(B_ * N_ / 16), dim3(32), 0, stream>>>(x, lnw, lnb, Wq, q16);
  k_qk_norm<<<dim3(16), dim3(256), 0, stream>>>(q16, invn);
  k_kvprojp<<<dim3(1024), dim3(32), 0, stream>>>(q16, WEF, kvFp);
  k_kvpack<<<dim3(64), dim3(256), 0, stream>>>(kvFp, bEF, kprojT, vproj);
  k_gramp<<<dim3(256), dim3(32), 0, stream>>>(q16, gramp);
  k_attnfin<<<dim3(8), dim3(32), 0, stream>>>(gramp, invn, temp1, attnca);
  k_xca<<<dim3(B_ * N_ / 256), dim3(256), 0, stream>>>(q16, attnca, xca);
  k_sa<<<dim3(B_ * NH_ * (N_ / 16)), dim3(32), 0, stream>>>(q16, invn, kprojT, vproj,
                                                            temp2, xsa);
  k_oproj<<<dim3(B_ * N_ / 16), dim3(32), 0, stream>>>(x, gamma, xsa, xca, Wo1, bo1,
                                                       Wo2, bo2, askip, askip16);
  k_wprep<<<dim3((2 * 64 * 64 * 27 + 255) / 256), dim3(256), 0, stream>>>(c1w, c2w, w16c);
  k_conv3<<<dim3(B_ * 32 * 32), dim3(32), 0, stream>>>(askip16, w16c, bn1w, bn1b,
                                                       nullptr, out1);
  k_conv3<<<dim3(B_ * 32 * 32), dim3(32), 0, stream>>>(out1, w16c + (size_t)27 * 64 * 64,
                                                       bn2w, bn2b, askip, out2);
  k_conv8<<<dim3(B_ * (N_ / 16)), dim3(32), 0, stream>>>(out2, c8w, c8b, askip, outp);
}
